// SimpleSTGNN_17437567222540
// MI455X (gfx1250) — matrix operand order repaired and checked by reading, not compiled, not run on hardware
//
#include <hip/hip_runtime.h>
#include <hip/hip_bf16.h>

// ---------------- problem constants (match reference) ----------------
#define NN   50000      // N_NODES (== M, divisible by 16)
#define FIN  16
#define H    64
#define NGCN 2
#define T    12
#define NE   1600000

typedef __attribute__((ext_vector_type(16))) _Float16 v16h;
typedef __attribute__((ext_vector_type(8)))  float    v8f;

__device__ __forceinline__ void atom_add(float* p, float v) {
  __hip_atomic_fetch_add(p, v, __ATOMIC_RELAXED, __HIP_MEMORY_SCOPE_AGENT);
}

// ---------------- generic WMMA GEMM ----------------------------------
// C = act(A @ B(ᵀ) [+bias1+bias2] [+C])
// A: [M,KK] f32 row-major.  B: TRANSB ? [N,KK] : [KK,N] f32 row-major.
// One wave computes MT consecutive 16-row tiles x one 16-col tile, so each
// B fragment is loaded once and reused by MT v_wmma ops (independent
// accumulators -> compiler can pipeline past the WMMA->WMMA hazard).
// KK is a compile-time constant: for KK%32==0 every fragment load is
// branch-free straight-line code; for KK==16 the A upper half is a
// compile-time zero and the B half-mask is a clamped-address + cndmask
// (no EXEC manipulation -> WMMA sees EXEC all-ones).
// flags: bit0 = accumulate into existing C, bit1 = ReLU.
template <int KK, bool TRANSB, int MT>
__global__ __launch_bounds__(256) void gemm_wmma(
    const float* __restrict__ A, int lda,
    const float* __restrict__ B, int ldb,
    const float* __restrict__ b1, const float* __restrict__ b2,
    float* __restrict__ C, int ldc, int M, int N, int flags)
{
  const int Nt = N >> 4;
  const int Mg = (((M + 15) >> 4) + MT - 1) / MT;
  const long wid = (long)blockIdx.x * 8 + (threadIdx.x >> 5);
  const int mg = (int)(wid / Nt);
  if (mg >= Mg) return;                       // wave-uniform exit
  const int nt = (int)(wid % Nt);
  const int m0 = mg * (MT * 16), n0 = nt << 4;
  const int lane = threadIdx.x & 31;
  const int r = lane & 15;                    // A-row / B-col within tile
  const int h = lane >> 4;                    // half-select per ISA layout

  v8f acc[MT] = {};

#pragma unroll
  for (int k0 = 0; k0 < KK; k0 += 32) {
    // ---- B fragment: 32x16 f16. lane(0-15)=col, elem e: K = k0 + h*8 + e (+8 if e>=8)
    v16h bf;
    if constexpr ((KK % 32) == 0) {
      const int kb = k0 + h * 8;              // statically in range
      if constexpr (TRANSB) {                 // B[N,K]: two runs of 8 floats (kb, kb+16)
        const float* Bp = B + (long)(n0 + r) * ldb + kb;
        float4 q0 = *(const float4*)(Bp);
        float4 q1 = *(const float4*)(Bp + 4);
        float4 q2 = *(const float4*)(Bp + 16);
        float4 q3 = *(const float4*)(Bp + 20);
        bf[0]=(_Float16)q0.x;  bf[1]=(_Float16)q0.y;  bf[2]=(_Float16)q0.z;  bf[3]=(_Float16)q0.w;
        bf[4]=(_Float16)q1.x;  bf[5]=(_Float16)q1.y;  bf[6]=(_Float16)q1.z;  bf[7]=(_Float16)q1.w;
        bf[8]=(_Float16)q2.x;  bf[9]=(_Float16)q2.y;  bf[10]=(_Float16)q2.z; bf[11]=(_Float16)q2.w;
        bf[12]=(_Float16)q3.x; bf[13]=(_Float16)q3.y; bf[14]=(_Float16)q3.z; bf[15]=(_Float16)q3.w;
      } else {                                // B[K,N]: coalesced across lanes
#pragma unroll
        for (int e = 0; e < 16; ++e)
          bf[e] = (_Float16)B[(long)(kb + e + (e >= 8 ? 8 : 0)) * ldb + n0 + r];
      }
    } else {
      // KK==16: elements e>=8 (K in [16,32)) are out of range -> clamp address, mask value.
      const int kb = k0 + h * 8;
#pragma unroll
      for (int e = 0; e < 16; ++e) {
        const int kk = kb + e + (e >= 8 ? 8 : 0);
        const int kc = (kk < KK) ? kk : 0;
        const float msk = (kk < KK) ? 1.f : 0.f;
        float v = TRANSB ? B[(long)(n0 + r) * ldb + kc]
                         : B[(long)kc * ldb + n0 + r];
        bf[e] = (_Float16)(v * msk);
      }
    }

    // ---- A fragments (one per m-tile) + WMMA, reusing bf
#pragma unroll
    for (int i = 0; i < MT; ++i) {
      int arow = m0 + i * 16 + r;
      arow = (arow < M) ? arow : (M - 1);
      const float* Arow = A + (long)arow * lda;
      v16h af;
      {
        const int kb = k0 + h * 8;            // always < KK (KK>=16)
        float4 p0 = *(const float4*)(Arow + kb);
        float4 p1 = *(const float4*)(Arow + kb + 4);
        af[0]=(_Float16)p0.x; af[1]=(_Float16)p0.y; af[2]=(_Float16)p0.z; af[3]=(_Float16)p0.w;
        af[4]=(_Float16)p1.x; af[5]=(_Float16)p1.y; af[6]=(_Float16)p1.z; af[7]=(_Float16)p1.w;
      }
      if constexpr ((KK % 32) == 0) {
        const int kb2 = k0 + 16 + h * 8;      // statically in range
        float4 p2 = *(const float4*)(Arow + kb2);
        float4 p3 = *(const float4*)(Arow + kb2 + 4);
        af[8]=(_Float16)p2.x;  af[9]=(_Float16)p2.y;  af[10]=(_Float16)p2.z; af[11]=(_Float16)p2.w;
        af[12]=(_Float16)p3.x; af[13]=(_Float16)p3.y; af[14]=(_Float16)p3.z; af[15]=(_Float16)p3.w;
      } else {
#pragma unroll
        for (int e = 8; e < 16; ++e) af[e] = (_Float16)0.f;  // K in [16,32): zero pad
      }
      acc[i] = __builtin_amdgcn_wmma_f32_16x16x32_f16(false, af, false, bf,
                                                      (short)0, acc[i], false, false);
    }
  }

  // ---- epilogue: bias / accumulate / relu / store
  float bias = 0.f;
  if (b1) bias += b1[n0 + r];
  if (b2) bias += b2[n0 + r];
#pragma unroll
  for (int i = 0; i < MT; ++i) {
#pragma unroll
    for (int rr = 0; rr < 8; ++rr) {          // D layout: row = rr + 8*h
      int row = m0 + i * 16 + rr + 8 * h;
      if (row < M) {
        long idx = (long)row * ldc + n0 + r;
        float v = acc[i][rr] + bias;
        if (flags & 1) v += C[idx];
        if (flags & 2) v = fmaxf(v, 0.f);
        C[idx] = v;
      }
    }
  }
}

template <int KK, bool TRANSB>
static void launch_gemm(const float* A, int lda, const float* B, int ldb,
                        const float* b1, const float* b2, float* C, int ldc,
                        int M, int N, int flags, hipStream_t stream) {
  constexpr int MT = 4;
  int Mg = (((M + 15) >> 4) + MT - 1) / MT;
  long waves = (long)Mg * (N >> 4);
  int blocks = (int)((waves + 7) / 8);
  gemm_wmma<KK, TRANSB, MT><<<blocks, 256, 0, stream>>>(
      A, lda, B, ldb, b1, b2, C, ldc, M, N, flags);
}

// ---------------- small helper kernels ----------------
__global__ void zero_f32(float* __restrict__ p, long n) {
  long i = (long)blockIdx.x * blockDim.x + threadIdx.x;
  if (i < n) p[i] = 0.f;
}

__global__ void deg_scatter(const int* __restrict__ dst, const float* __restrict__ w,
                            float* __restrict__ deg, int E) {
  int e = blockIdx.x * blockDim.x + threadIdx.x;
  if (e < E) atom_add(&deg[dst[e]], w[e]);
}

__global__ void deg_finalize(const float* __restrict__ deg, float* __restrict__ dinv,
                             float* __restrict__ snrm, int n) {
  int i = blockIdx.x * blockDim.x + threadIdx.x;
  if (i < n) {
    float d = deg[i] + 1.0f;
    dinv[i] = rsqrtf(d);
    snrm[i] = 1.0f / d;          // dinv^2
  }
}

__global__ void enorm_kernel(const int* __restrict__ src, const int* __restrict__ dst,
                             const float* __restrict__ w, const float* __restrict__ dinv,
                             float* __restrict__ enrm, int E) {
  int e = blockIdx.x * blockDim.x + threadIdx.x;
  if (e < E) enrm[e] = dinv[src[e]] * dinv[dst[e]] * w[e];
}

// agg[m,j] = self_norm[m]*zw[m,j] + gcn_b[j]
__global__ void agg_init(const float* __restrict__ zw, const float* __restrict__ snrm,
                         const float* __restrict__ gb, float* __restrict__ agg, long n) {
  long i = (long)blockIdx.x * blockDim.x + threadIdx.x;
  if (i < n) {
    long m = i >> 6; int j = (int)(i & 63);
    agg[i] = snrm[m] * zw[i] + gb[j];
  }
}

// wave-per-edge scatter: agg[dst] += enorm[e] * zw[src]  (L2-resident f32 atomics)
__global__ __launch_bounds__(256) void edge_scatter(
    const int* __restrict__ src, const int* __restrict__ dst,
    const float* __restrict__ enrm, const float* __restrict__ zw,
    float* __restrict__ agg, int E)
{
  int e = blockIdx.x * 8 + (threadIdx.x >> 5);
  if (e >= E) return;
  int lane = threadIdx.x & 31;
  int s = src[e], d = dst[e];
  float w = enrm[e];
  float2 v = *(const float2*)(zw + (long)s * H + lane * 2);
  float* ap = agg + (long)d * H + lane * 2;
  atom_add(ap, v.x * w);
  atom_add(ap + 1, v.y * w);
}

// wave-per-node LayerNorm(64) + ReLU (wave32 shuffle reduction)
__global__ __launch_bounds__(256) void ln_relu(
    const float* __restrict__ agg, const float* __restrict__ g,
    const float* __restrict__ b, float* __restrict__ z, int M)
{
  int m = blockIdx.x * 8 + (threadIdx.x >> 5);
  if (m >= M) return;
  int lane = threadIdx.x & 31;
  const float* row = agg + (long)m * H;
  float v0 = row[lane], v1 = row[lane + 32];
  float s = v0 + v1;
#pragma unroll
  for (int o = 16; o > 0; o >>= 1) s += __shfl_xor(s, o, 32);
  float mu = s * (1.f / 64.f);
  float d0 = v0 - mu, d1 = v1 - mu;
  float vs = d0 * d0 + d1 * d1;
#pragma unroll
  for (int o = 16; o > 0; o >>= 1) vs += __shfl_xor(vs, o, 32);
  float rstd = rsqrtf(vs * (1.f / 64.f) + 1e-5f);
  float* out = z + (long)m * H;
  out[lane]      = fmaxf(d0 * rstd * g[lane]      + b[lane],      0.f);
  out[lane + 32] = fmaxf(d1 * rstd * g[lane + 32] + b[lane + 32], 0.f);
}

// LSTM cell pointwise: gates [M,256] order i,f,g,o ; updates h,c in place
__global__ void lstm_cell(const float* __restrict__ gates,
                          float* __restrict__ hh, float* __restrict__ cc, long n) {
  long i = (long)blockIdx.x * blockDim.x + threadIdx.x;
  if (i >= n) return;
  long m = i >> 6; int j = (int)(i & 63);
  const float* gr = gates + m * 256;
  float gi = gr[j], gf = gr[64 + j], gg = gr[128 + j], go = gr[192 + j];
  float si = 1.f / (1.f + expf(-gi));
  float sf = 1.f / (1.f + expf(-gf));
  float so = 1.f / (1.f + expf(-go));
  float cn = sf * cc[i] + si * tanhf(gg);
  cc[i] = cn;
  hh[i] = so * tanhf(cn);
}

__global__ void fc2_kernel(const float* __restrict__ a, const float* __restrict__ w,
                           const float* __restrict__ b, float* __restrict__ out, int M) {
  int m = blockIdx.x * blockDim.x + threadIdx.x;
  if (m >= M) return;
  const float* ar = a + (long)m * 32;
  float s = b[0];
#pragma unroll
  for (int j = 0; j < 32; ++j) s += ar[j] * w[j];
  out[m] = s;
}

// ---------------- host orchestration ----------------
extern "C" void kernel_launch(void* const* d_in, const int* in_sizes, int n_in,
                              void* d_out, int out_size, void* d_ws, size_t ws_size,
                              hipStream_t stream)
{
  const float* x     = (const float*)d_in[0];
  const int*   ei    = (const int*)  d_in[1];
  const float* ew    = (const float*)d_in[2];
  const float* W_in  = (const float*)d_in[3];
  const float* b_in  = (const float*)d_in[4];
  const float* gcnW  = (const float*)d_in[5];
  const float* gcnB  = (const float*)d_in[6];
  const float* lnG   = (const float*)d_in[7];
  const float* lnB   = (const float*)d_in[8];
  const float* Wih0  = (const float*)d_in[9];
  const float* Whh0  = (const float*)d_in[10];
  const float* bih0  = (const float*)d_in[11];
  const float* bhh0  = (const float*)d_in[12];
  const float* Wih1  = (const float*)d_in[13];
  const float* Whh1  = (const float*)d_in[14];
  const float* bih1  = (const float*)d_in[15];
  const float* bhh1  = (const float*)d_in[16];
  const float* fc1W  = (const float*)d_in[17];
  const float* fc1b  = (const float*)d_in[18];
  const float* fc2W  = (const float*)d_in[19];
  const float* fc2b  = (const float*)d_in[20];
  float* out = (float*)d_out;

  const int* src = ei;
  const int* dst = ei + NE;

  // carve workspace (float units, 256B-aligned chunks)
  float* p = (float*)d_ws;
  auto carve = [&](size_t n) { float* q = p; p += (n + 63) & ~(size_t)63; return q; };
  float* deg   = carve(NN);
  float* dinv  = carve(NN);
  float* snrm  = carve(NN);
  float* enrm  = carve(NE);
  float* z     = carve((size_t)NN * H);
  float* zw    = carve((size_t)NN * H);
  float* agg   = carve((size_t)NN * H);
  float* hc    = carve((size_t)NN * H * 4);   // h0,c0,h1,c1 contiguous
  float* h0 = hc;
  float* c0 = hc + (size_t)NN * H;
  float* h1 = hc + (size_t)NN * H * 2;
  float* c1 = hc + (size_t)NN * H * 3;
  float* gates = carve((size_t)NN * 4 * H);
  float* f1o   = carve((size_t)NN * 32);

  // ---- per-call state init (deterministic across graph replays) ----
  zero_f32<<<(NN + 255) / 256, 256, 0, stream>>>(deg, NN);
  zero_f32<<<(int)(((long)NN * H * 4 + 255) / 256), 256, 0, stream>>>(hc, (long)NN * H * 4);

  // ---- GCN normalization precompute ----
  deg_scatter<<<(NE + 255) / 256, 256, 0, stream>>>(dst, ew, deg, NE);
  deg_finalize<<<(NN + 255) / 256, 256, 0, stream>>>(deg, dinv, snrm, NN);
  enorm_kernel<<<(NE + 255) / 256, 256, 0, stream>>>(src, dst, ew, dinv, enrm, NE);

  const long nMH = (long)NN * H;
  for (int t = 0; t < T; ++t) {
    const float* xt = x + (long)t * NN * FIN;
    // input projection: z = relu(x_t @ W_in + b_in)   [M,16]@[16,64]
    launch_gemm<FIN, false>(xt, FIN, W_in, H, b_in, nullptr, z, H, NN, H, 2, stream);

    // two GCN layers
    for (int l = 0; l < NGCN; ++l) {
      launch_gemm<H, false>(z, H, gcnW + (long)l * H * H, H, nullptr, nullptr,
                            zw, H, NN, H, 0, stream);
      agg_init<<<(int)((nMH + 255) / 256), 256, 0, stream>>>(zw, snrm, gcnB + l * H, agg, nMH);
      edge_scatter<<<(NE + 7) / 8, 256, 0, stream>>>(src, dst, enrm, zw, agg, NE);
      ln_relu<<<(NN + 7) / 8, 256, 0, stream>>>(agg, lnG + l * H, lnB + l * H, z, NN);
    }

    // LSTM layer 0: gates = z@Wih0^T + h0@Whh0^T + bih0 + bhh0
    launch_gemm<H, true>(z, H, Wih0, H, bih0, bhh0, gates, 4 * H, NN, 4 * H, 0, stream);
    launch_gemm<H, true>(h0, H, Whh0, H, nullptr, nullptr, gates, 4 * H, NN, 4 * H, 1, stream);
    lstm_cell<<<(int)((nMH + 255) / 256), 256, 0, stream>>>(gates, h0, c0, nMH);

    // LSTM layer 1: input h0
    launch_gemm<H, true>(h0, H, Wih1, H, bih1, bhh1, gates, 4 * H, NN, 4 * H, 0, stream);
    launch_gemm<H, true>(h1, H, Whh1, H, nullptr, nullptr, gates, 4 * H, NN, 4 * H, 1, stream);
    lstm_cell<<<(int)((nMH + 255) / 256), 256, 0, stream>>>(gates, h1, c1, nMH);
  }

  // head: out = relu(h1 @ fc1_W + fc1_b) @ fc2_W + fc2_b
  launch_gemm<H, false>(h1, H, fc1W, 32, fc1b, nullptr, f1o, 32, NN, 32, 2, stream);
  fc2_kernel<<<(NN + 255) / 256, 256, 0, stream>>>(f1o, fc2W, fc2b, out, NN);
}
